// CentroidHead_69200513073483
// MI455X (gfx1250) — matrix operand order repaired and checked by reading, not compiled, not run on hardware
//
#include <hip/hip_runtime.h>

#define NPTS   262144
#define DFEAT  128
#define NB     2
#define SS     192
#define GRIDP  194                       // S+2 padded
#define GRIDN  (2*GRIDP*GRIDP*GRIDP)     // 14,602,768 cells
#define KTOP   128
#define DDIM   16
#define MAXCAND 32768
#define SCORE_THR 0.1f
#define CENT_THR  0.2f

typedef __attribute__((ext_vector_type(16))) __bf16 bf16x16;
typedef __attribute__((ext_vector_type(8)))  float  f32x8;
typedef __attribute__((ext_vector_type(4)))  unsigned u32x4;
typedef __attribute__((ext_vector_type(8)))  int      i32x8;
typedef __attribute__((ext_vector_type(4)))  int      i32x4;

union Frag { bf16x16 v; unsigned u[8]; };

__device__ __forceinline__ unsigned pack_bf2(float a, float b) {
  unsigned ua = __float_as_uint(a), ub = __float_as_uint(b);
  ua = (ua + 0x7FFFu + ((ua >> 16) & 1u)) >> 16;          // round-to-nearest-even
  ub = (ub + 0x7FFFu + ((ub >> 16) & 1u)) >> 16;
  return ua | (ub << 16);
}
__device__ __forceinline__ unsigned short f2bf(float a) {
  unsigned ua = __float_as_uint(a);
  return (unsigned short)((ua + 0x7FFFu + ((ua >> 16) & 1u)) >> 16);
}
__device__ __forceinline__ float bf2f(unsigned short h) {
  return __uint_as_float(((unsigned)h) << 16);
}

// ---------- score kernel LDS partition (bytes) ----------
//  [0      , 65536)  : shXf  128x128 f32 X tile (TDM dest); H / h1 written
//                      bf16 in-place per-stripe afterwards (u16 stride 256)
//  [65536  , 98304)  : shW0T W0 transposed [n][kpair] (8192 u32)
//  [98304  , 114688) : shW1T W1 transposed [n][kpair] (4096 u32)
//  [114688 , 115200) : shB0 128 f32
//  [115200 , 115456) : shB1 64 f32
//  [115456 , 115712) : shW2f 64 f32
#define SM_SCORE 115712
// ---------- descriptor kernel LDS partition (round-1 layout) ----------
#define SM_DESC  93248

// ---------------- Kernel 1: scoring MLP (N rows) ----------------
__global__ __launch_bounds__(256) void score_mlp_kernel(
    const float* __restrict__ feats, const float* __restrict__ aux,
    const float* __restrict__ sw0, const float* __restrict__ sb0,
    const float* __restrict__ sw1, const float* __restrict__ sb1,
    const float* __restrict__ sw2, const float* __restrict__ sb2,
    float* __restrict__ scores) {
  extern __shared__ char smem[];
  float*    shXf  = (float*)smem;                       // fp32 X tile
  unsigned* shW0T = (unsigned*)(smem + 65536);
  unsigned* shW1T = (unsigned*)(smem + 98304);
  float*    shB0  = (float*)(smem + 114688);
  float*    shB1  = (float*)(smem + 115200);
  float*    shW2f = (float*)(smem + 115456);

  const int t = threadIdx.x;
  const int lane = t & 31, wave = t >> 5;
  const int half = lane >> 4, ln = lane & 15;
  const long rb = (long)blockIdx.x * 128;

  // ---- TDM: DMA the 128x128 f32 tile feats[rb..rb+128) into LDS ----
  if (wave == 0) {
    unsigned long long ga = (unsigned long long)(uintptr_t)(feats + rb * DFEAT);
    unsigned ldsoff = (unsigned)(uintptr_t)(void*)shXf;   // low 32 bits = LDS offset
    u32x4 g0;
    g0[0] = 1u;                                           // count=1 (valid user D#)
    g0[1] = ldsoff;                                       // lds_addr
    g0[2] = (unsigned)(ga & 0xFFFFFFFFull);               // global_addr[31:0]
    g0[3] = (unsigned)((ga >> 32) & 0x01FFFFFFull)        // global_addr[56:32]
            | (2u << 30);                                 // type = 2 ("image")
    i32x8 g1;
    g1[0] = (int)(2u << 16);       // data_size = 2 (4 bytes); no multicast/pad
    g1[1] = (int)(128u << 16);     // tensor_dim0 = 128 (low 16 bits at [63:48])
    g1[2] = (int)(128u << 16);     // tensor_dim0 hi = 0 | tensor_dim1 = 128
    g1[3] = (int)(128u << 16);     // tensor_dim1 hi = 0 | tile_dim0 = 128
    g1[4] = 128;                   // tile_dim1 = 128, tile_dim2 = 0
    g1[5] = 128;                   // tensor_dim0_stride = 128 (low 32)
    g1[6] = 0;                     // stride0 hi | tensor_dim1_stride lo
    g1[7] = 0;
    i32x4 z4 = {0, 0, 0, 0};       // groups 2/3 unused (2D tensor)
    i32x8 z8 = {0, 0, 0, 0, 0, 0, 0, 0};
    __builtin_amdgcn_tensor_load_to_lds(g0, g1, z4, z4, z8, 0);
  }

  // ---- stage weights (all waves, overlaps the TDM transfer) ----
  for (int p = t; p < 8192; p += 256) {
    int n = p >> 6, kp = p & 63;
    shW0T[n * 64 + kp] = pack_bf2(sw0[(2 * kp) * 128 + n], sw0[(2 * kp + 1) * 128 + n]);
  }
  for (int p = t; p < 4096; p += 256) {
    int n = p >> 6, kp = p & 63;
    shW1T[n * 64 + kp] = pack_bf2(sw1[(2 * kp) * 64 + n], sw1[(2 * kp + 1) * 64 + n]);
  }
  if (t < 128) shB0[t] = sb0[t];
  if (t < 64) { shB1[t] = sb1[t]; shW2f[t] = sw2[t]; }

  if (wave == 0) __builtin_amdgcn_s_wait_tensorcnt(0);
  __syncthreads();

  const int row = wave * 16 + ln;
  unsigned short* shH  = (unsigned short*)shXf;   // bf16 overlay, stride 256/row
  unsigned*       shH32 = (unsigned*)shXf;

  // ---- layer 0: X(16x128) @ W0(128x128), bf16 WMMA fp32 accum ----
  f32x8 acc[8];
  #pragma unroll
  for (int nt = 0; nt < 8; ++nt) {
    float b = shB0[nt * 16 + ln];
    #pragma unroll
    for (int r = 0; r < 8; ++r) acc[nt][r] = b;
  }
  const float2* xr = (const float2*)(shXf + row * 128);
  #pragma unroll
  for (int kc = 0; kc < 4; ++kc) {
    Frag A;   // convert fp32 pairs -> bf16 at fragment build time
    #pragma unroll
    for (int v = 0; v < 8; ++v) {
      int poff = half * 8 + v;
      float2 xv = xr[kc * 16 + poff];
      A.u[v] = pack_bf2(xv.x, xv.y);
    }
    #pragma unroll
    for (int nt = 0; nt < 8; ++nt) {
      Frag Bf;
      int n = nt * 16 + ln;
      #pragma unroll
      for (int v = 0; v < 8; ++v)
        Bf.u[v] = shW0T[n * 64 + kc * 16 + half * 8 + v];
      acc[nt] = __builtin_amdgcn_wmma_f32_16x16x32_bf16(
          false, A.v, false, Bf.v, (short)0, acc[nt], false, false);
    }
  }
  // relu + store H bf16 in place over own fp32 stripe (all reads done)
  #pragma unroll
  for (int nt = 0; nt < 8; ++nt)
    #pragma unroll
    for (int r = 0; r < 8; ++r) {
      int m = wave * 16 + r + half * 8;
      shH[m * 256 + nt * 16 + ln] = f2bf(fmaxf(acc[nt][r], 0.0f));
    }

  // ---- layer 1: H(16x128) @ W1(128x64) ----
  f32x8 acc1[4];
  #pragma unroll
  for (int nt = 0; nt < 4; ++nt) {
    float b = shB1[nt * 16 + ln];
    #pragma unroll
    for (int r = 0; r < 8; ++r) acc1[nt][r] = b;
  }
  #pragma unroll
  for (int kc = 0; kc < 4; ++kc) {
    Frag A;
    #pragma unroll
    for (int v = 0; v < 8; ++v) {
      int poff = half * 8 + v;
      A.u[v] = shH32[row * 128 + kc * 16 + poff];
    }
    #pragma unroll
    for (int nt = 0; nt < 4; ++nt) {
      Frag Bf;
      int n = nt * 16 + ln;
      #pragma unroll
      for (int v = 0; v < 8; ++v)
        Bf.u[v] = shW1T[n * 64 + kc * 16 + half * 8 + v];
      acc1[nt] = __builtin_amdgcn_wmma_f32_16x16x32_bf16(
          false, A.v, false, Bf.v, (short)0, acc1[nt], false, false);
    }
  }
  // relu + store h1 bf16 into cols [0,64) of own stripe rows
  #pragma unroll
  for (int nt = 0; nt < 4; ++nt)
    #pragma unroll
    for (int r = 0; r < 8; ++r) {
      int m = wave * 16 + r + half * 8;
      shH[m * 256 + nt * 16 + ln] = f2bf(fmaxf(acc1[nt][r], 0.0f));
    }

  // ---- layer 2: 64 -> 1 (VALU), lane halves split K then combine ----
  float part = 0.0f;
  #pragma unroll
  for (int k = 0; k < 32; ++k)
    part += bf2f(shH[row * 256 + half * 32 + k]) * shW2f[half * 32 + k];
  part += __shfl_xor(part, 16, 32);
  if (half == 0) {
    float logit = part + sb2[0] + aux[rb + row];
    scores[rb + row] = 1.0f / (1.0f + expf(-logit));
  }
}

// ---------------- Kernel 2: grid init ----------------
__global__ void grid_init_kernel(int* __restrict__ grid, int* __restrict__ counter) {
  long i = (long)blockIdx.x * blockDim.x + threadIdx.x;
  long stride = (long)gridDim.x * blockDim.x;
  for (; i < GRIDN; i += stride) grid[i] = 0xFF800000;  // -inf bits
  if (blockIdx.x == 0 && threadIdx.x == 0) *counter = 0;
}

// ---------------- Kernel 3: scatter-max into grid ----------------
__global__ __launch_bounds__(256) void scatter_kernel(
    const float* __restrict__ scores, const int* __restrict__ coords,
    int* __restrict__ grid) {
  int i = blockIdx.x * blockDim.x + threadIdx.x;
  if (i >= NPTS) return;
  float s = scores[i];
  if (s > SCORE_THR) {
    int4 c = ((const int4*)coords)[i];
    int x = c.y + 1, y = c.z + 1, z = c.w + 1;
    long cell = (((long)c.x * GRIDP + x) * GRIDP + y) * GRIDP + z;
    atomicMax(&grid[cell], __float_as_int(s));   // positive floats: int max == float max
  }
}

// ---------------- Kernel 4: peak detect + candidate compaction ----------------
__global__ __launch_bounds__(256) void peaks_kernel(
    const float* __restrict__ scores, const int* __restrict__ coords,
    const int* __restrict__ grid, int* __restrict__ counter,
    float* __restrict__ cand_s, int* __restrict__ cand_i) {
  int i = blockIdx.x * blockDim.x + threadIdx.x;
  if (i >= NPTS) return;
  float s = scores[i];
  if (s <= CENT_THR) return;          // implies active (CENT_THR > SCORE_THR)
  int4 c = ((const int4*)coords)[i];
  int x = c.y + 1, y = c.z + 1, z = c.w + 1;
  const int* g = grid + ((((long)c.x * GRIDP + x) * GRIDP + y) * GRIDP + z);
  int hmax = (int)0x80000000;
  #pragma unroll
  for (int dx = -1; dx <= 1; ++dx)
    #pragma unroll
    for (int dy = -1; dy <= 1; ++dy)
      #pragma unroll
      for (int dz = -1; dz <= 1; ++dz) {
        int v = g[(dx * GRIDP + dy) * GRIDP + dz];
        hmax = max(hmax, v);
      }
  if (hmax == __float_as_int(s)) {
    int pos = atomicAdd(counter, 1);
    if (pos < MAXCAND) { cand_s[pos] = s; cand_i[pos] = i; }
  }
}

// ---------------- Kernel 5: deterministic top-K selection ----------------
__global__ __launch_bounds__(256) void topk_select_kernel(
    const int* __restrict__ counter, float* __restrict__ cand_s,
    const int* __restrict__ cand_i, int* __restrict__ topIdx,
    float* __restrict__ confW) {
  __shared__ unsigned long long sk[256];
  const int tid = threadIdx.x;
  int n = *counter;
  if (n > MAXCAND) n = MAXCAND;
  for (int k = 0; k < KTOP; ++k) {
    unsigned long long best = 0ull;
    for (int c = tid; c < n; c += 256) {
      float s = cand_s[c];
      if (s > 0.0f) {   // taken entries are marked -1
        unsigned long long key =
            ((unsigned long long)__float_as_uint(s) << 32) |
            (unsigned)(0xFFFFFFFFu - (unsigned)cand_i[c]);  // tie: smaller point idx wins
        if (key > best) best = key;
      }
    }
    sk[tid] = best;
    __syncthreads();
    for (int off = 128; off > 0; off >>= 1) {
      if (tid < off && sk[tid + off] > sk[tid]) sk[tid] = sk[tid + off];
      __syncthreads();
    }
    unsigned long long bk = sk[0];
    if (bk != 0ull) {
      int bi = (int)(0xFFFFFFFFu - (unsigned)(bk & 0xFFFFFFFFull));
      if (tid == 0) {
        topIdx[k] = bi;
        confW[k] = __uint_as_float((unsigned)(bk >> 32));
      }
      for (int c = tid; c < n; c += 256)
        if (cand_i[c] == bi) cand_s[c] = -1.0f;
    } else if (tid == 0) {
      topIdx[k] = -1;
      confW[k] = 0.0f;
    }
    __syncthreads();
  }
}

// ---------------- Kernel 6: descriptor MLP on top-K rows ----------------
__global__ __launch_bounds__(256) void desc_mlp_kernel(
    const float* __restrict__ feats,
    const float* __restrict__ dw0, const float* __restrict__ db0,
    const float* __restrict__ dw1, const float* __restrict__ db1,
    const float* __restrict__ dw2, const float* __restrict__ db2,
    const int* __restrict__ topIdx, const float* __restrict__ confW,
    float* __restrict__ outDesc, float* __restrict__ outConf) {
  extern __shared__ char smem[];
  unsigned short* shXH = (unsigned short*)smem;        // 128x128 bf16 (stride 128)
  unsigned* shW0T = (unsigned*)(smem + 32768);
  unsigned* shW1T = (unsigned*)(smem + 65536);
  float* shB0 = (float*)(smem + 81920);
  float* shB1 = (float*)(smem + 82432);
  unsigned* shW2T = (unsigned*)(smem + 82944);
  float* shB2 = (float*)(smem + 84992);
  float* shD = (float*)(smem + 85056);

  const int t = threadIdx.x;
  unsigned* shX32 = (unsigned*)shXH;
  // gather 128 rows of feats (zeros for invalid slots)
  for (int p = t; p < 8192; p += 256) {
    int r = p >> 6, kp = p & 63;
    int idx = topIdx[r];
    float2 xv;
    if (idx >= 0) xv = *(const float2*)(feats + (long)idx * DFEAT + kp * 2);
    else { xv.x = 0.0f; xv.y = 0.0f; }
    shX32[p] = pack_bf2(xv.x, xv.y);
  }
  for (int p = t; p < 8192; p += 256) {
    int n = p >> 6, kp = p & 63;
    shW0T[n * 64 + kp] = pack_bf2(dw0[(2 * kp) * 128 + n], dw0[(2 * kp + 1) * 128 + n]);
  }
  for (int p = t; p < 4096; p += 256) {
    int n = p >> 6, kp = p & 63;
    shW1T[n * 64 + kp] = pack_bf2(dw1[(2 * kp) * 64 + n], dw1[(2 * kp + 1) * 64 + n]);
  }
  for (int p = t; p < 512; p += 256) {
    int n = p >> 5, kp = p & 31;
    shW2T[n * 32 + kp] = pack_bf2(dw2[(2 * kp) * DDIM + n], dw2[(2 * kp + 1) * DDIM + n]);
  }
  if (t < 128) shB0[t] = db0[t];
  if (t < 64) shB1[t] = db1[t];
  if (t < 16) shB2[t] = db2[t];
  __syncthreads();

  const int lane = t & 31, wave = t >> 5;
  const int half = lane >> 4, ln = lane & 15;
  const int row = wave * 16 + ln;
  unsigned* shH32 = (unsigned*)shXH;

  // ---- layer 0: 128 -> 128 ----
  f32x8 acc[8];
  #pragma unroll
  for (int nt = 0; nt < 8; ++nt) {
    float b = shB0[nt * 16 + ln];
    #pragma unroll
    for (int r = 0; r < 8; ++r) acc[nt][r] = b;
  }
  #pragma unroll
  for (int kc = 0; kc < 4; ++kc) {
    Frag A;
    #pragma unroll
    for (int v = 0; v < 8; ++v) {
      int poff = half * 8 + v;
      A.u[v] = shH32[row * 64 + kc * 16 + poff];
    }
    #pragma unroll
    for (int nt = 0; nt < 8; ++nt) {
      Frag Bf;
      int n = nt * 16 + ln;
      #pragma unroll
      for (int v = 0; v < 8; ++v)
        Bf.u[v] = shW0T[n * 64 + kc * 16 + half * 8 + v];
      acc[nt] = __builtin_amdgcn_wmma_f32_16x16x32_bf16(
          false, A.v, false, Bf.v, (short)0, acc[nt], false, false);
    }
  }
  #pragma unroll
  for (int nt = 0; nt < 8; ++nt)
    #pragma unroll
    for (int r = 0; r < 8; ++r) {
      int m = wave * 16 + r + half * 8;
      shXH[m * 128 + nt * 16 + ln] = f2bf(fmaxf(acc[nt][r], 0.0f));
    }

  // ---- layer 1: 128 -> 64 ----
  f32x8 acc1[4];
  #pragma unroll
  for (int nt = 0; nt < 4; ++nt) {
    float b = shB1[nt * 16 + ln];
    #pragma unroll
    for (int r = 0; r < 8; ++r) acc1[nt][r] = b;
  }
  #pragma unroll
  for (int kc = 0; kc < 4; ++kc) {
    Frag A;
    #pragma unroll
    for (int v = 0; v < 8; ++v) {
      int poff = half * 8 + v;
      A.u[v] = shH32[row * 64 + kc * 16 + poff];
    }
    #pragma unroll
    for (int nt = 0; nt < 4; ++nt) {
      Frag Bf;
      int n = nt * 16 + ln;
      #pragma unroll
      for (int v = 0; v < 8; ++v)
        Bf.u[v] = shW1T[n * 64 + kc * 16 + half * 8 + v];
      acc1[nt] = __builtin_amdgcn_wmma_f32_16x16x32_bf16(
          false, A.v, false, Bf.v, (short)0, acc1[nt], false, false);
    }
  }
  #pragma unroll
  for (int nt = 0; nt < 4; ++nt)
    #pragma unroll
    for (int r = 0; r < 8; ++r) {
      int m = wave * 16 + r + half * 8;
      shXH[m * 128 + nt * 16 + ln] = f2bf(fmaxf(acc1[nt][r], 0.0f));
    }

  // ---- layer 2: h1(16x64) @ dw2(64x16) — single WMMA tile, K=64 ----
  f32x8 acc2;
  {
    float b = shB2[ln];
    #pragma unroll
    for (int r = 0; r < 8; ++r) acc2[r] = b;
  }
  #pragma unroll
  for (int kc = 0; kc < 2; ++kc) {
    Frag A;
    #pragma unroll
    for (int v = 0; v < 8; ++v) {
      int poff = half * 8 + v;
      A.u[v] = shH32[row * 64 + kc * 16 + poff];
    }
    Frag Bf;
    #pragma unroll
    for (int v = 0; v < 8; ++v)
      Bf.u[v] = shW2T[ln * 32 + kc * 16 + half * 8 + v];
    acc2 = __builtin_amdgcn_wmma_f32_16x16x32_bf16(
        false, A.v, false, Bf.v, (short)0, acc2, false, false);
  }
  #pragma unroll
  for (int r = 0; r < 8; ++r) {
    int m = wave * 16 + r + half * 8;
    shD[m * DDIM + ln] = acc2[r];
  }
  __syncthreads();

  // normalize rows, scale by conf
  if (t < KTOP) {
    float c = confW[t];
    float d[DDIM];
    float ss = 0.0f;
    #pragma unroll
    for (int j = 0; j < DDIM; ++j) { d[j] = shD[t * DDIM + j]; ss += d[j] * d[j]; }
    float scale = c / fmaxf(sqrtf(ss), 1e-12f);
    #pragma unroll
    for (int j = 0; j < DDIM; ++j) outDesc[t * DDIM + j] = d[j] * scale;
    outConf[t] = c;
  }
}

extern "C" void kernel_launch(void* const* d_in, const int* in_sizes, int n_in,
                              void* d_out, int out_size, void* d_ws, size_t ws_size,
                              hipStream_t stream) {
  const float* feats = (const float*)d_in[0];
  const float* aux   = (const float*)d_in[1];
  const int*   coords = (const int*)d_in[2];
  // d_in[3] = mask (unused by reference)
  const float* sw0 = (const float*)d_in[4];
  const float* sb0 = (const float*)d_in[5];
  const float* sw1 = (const float*)d_in[6];
  const float* sb1 = (const float*)d_in[7];
  const float* sw2 = (const float*)d_in[8];
  const float* sb2 = (const float*)d_in[9];
  const float* dw0 = (const float*)d_in[10];
  const float* db0 = (const float*)d_in[11];
  const float* dw1 = (const float*)d_in[12];
  const float* db1 = (const float*)d_in[13];
  const float* dw2 = (const float*)d_in[14];
  const float* db2 = (const float*)d_in[15];
  float* out = (float*)d_out;

  // workspace layout
  int*   grid    = (int*)d_ws;                 // GRIDN ints (~58.4 MB)
  int*   counter = grid + GRIDN;               // 1
  float* cand_s  = (float*)(counter + 1);      // MAXCAND
  int*   cand_i  = (int*)(cand_s + MAXCAND);   // MAXCAND
  int*   topIdx  = cand_i + MAXCAND;           // KTOP
  float* confW   = (float*)(topIdx + KTOP);    // KTOP

  grid_init_kernel<<<2048, 256, 0, stream>>>(grid, counter);
  score_mlp_kernel<<<NPTS / 128, 256, SM_SCORE, stream>>>(
      feats, aux, sw0, sb0, sw1, sb1, sw2, sb2, out);
  scatter_kernel<<<NPTS / 256, 256, 0, stream>>>(out, coords, grid);
  peaks_kernel<<<NPTS / 256, 256, 0, stream>>>(out, coords, grid, counter, cand_s, cand_i);
  topk_select_kernel<<<1, 256, 0, stream>>>(counter, cand_s, cand_i, topIdx, confW);
  desc_mlp_kernel<<<1, 256, SM_DESC, stream>>>(
      feats, dw0, db0, dw1, db1, dw2, db2, topIdx, confW,
      out + NPTS, out + NPTS + KTOP * DDIM);
}
